// W8A8B8O8Linear_24524263260388
// MI455X (gfx1250) — 8-bit matrix operand order repaired and checked by reading, compiled, not run on hardware
//
#include <hip/hip_runtime.h>
#include <stdint.h>


typedef int v8i __attribute__((ext_vector_type(8)));
typedef int v4i_n __attribute__((vector_size(16)));
typedef unsigned tdm_v4u __attribute__((vector_size(16)));
typedef int      tdm_v8i __attribute__((vector_size(32)));
typedef int      tdm_v4i __attribute__((vector_size(16)));
#define AS1 __attribute__((address_space(1)))
#define AS3 __attribute__((address_space(3)))

#define K_DIM   4096
#define N_DIM   4096
#define M_TOTAL 8192
#define BM      128
#define BN      128
#define BK      64
#define NK      (K_DIM / BK)
#define KPAD    80
#define XP_BYTES ((size_t)M_TOTAL * K_DIM)

#if defined(__gfx1250__) && __has_builtin(__builtin_amdgcn_tensor_load_to_lds)
#define USE_TDM 1
#else
#define USE_TDM 0
#endif
#if !USE_TDM && defined(__gfx1250__) && __has_builtin(__builtin_amdgcn_global_load_async_to_lds_b128)
#define USE_ASYNC_LDS 1
#else
#define USE_ASYNC_LDS 0
#endif

__device__ __forceinline__ unsigned pack_low4(unsigned a, unsigned b,
                                              unsigned c, unsigned d) {
    unsigned t_ab = __builtin_amdgcn_perm(b, a, 0x00000400u);
    unsigned t_cd = __builtin_amdgcn_perm(d, c, 0x00000400u);
    return __builtin_amdgcn_perm(t_cd, t_ab, 0x05040100u);
}

__device__ __forceinline__ unsigned lds_off(const void* p) {
    return (unsigned)(size_t)(AS3 const char*)p;
}

__global__ __launch_bounds__(256)
void pack_x_kernel(const int* __restrict__ x, signed char* __restrict__ xp) {
    const size_t i0 = ((size_t)blockIdx.x * 256 + threadIdx.x) * 16;
    const int4 v0 = *(const int4*)(x + i0);
    const int4 v1 = *(const int4*)(x + i0 + 4);
    const int4 v2 = *(const int4*)(x + i0 + 8);
    const int4 v3 = *(const int4*)(x + i0 + 12);
    int4 r;
    r.x = (int)pack_low4(v0.x, v0.y, v0.z, v0.w);
    r.y = (int)pack_low4(v1.x, v1.y, v1.z, v1.w);
    r.z = (int)pack_low4(v2.x, v2.y, v2.z, v2.w);
    r.w = (int)pack_low4(v3.x, v3.y, v3.z, v3.w);
    *(int4*)(xp + i0) = r;
}

__global__ __launch_bounds__(256)
void pack_wt_kernel(const int* __restrict__ w, signed char* __restrict__ wt) {
    const int n  = blockIdx.x * 256 + threadIdx.x;
    const int k0 = blockIdx.y * 16;
    int4 r;
#pragma unroll
    for (int q = 0; q < 4; ++q) {
        const int* wp = w + (size_t)(k0 + q * 4) * N_DIM + n;
        unsigned p = pack_low4((unsigned)wp[0],
                               (unsigned)wp[N_DIM],
                               (unsigned)wp[2 * N_DIM],
                               (unsigned)wp[3 * N_DIM]);
        ((unsigned*)&r)[q] = p;
    }
    *(int4*)(wt + (size_t)n * K_DIM + k0) = r;
}

#if USE_TDM
__device__ __forceinline__ void tdm_issue_tile(const signed char* gbase,
                                               unsigned lds_addr) {
    const unsigned long long ga = (unsigned long long)(size_t)gbase;
    tdm_v4u g0 = {
        1u,
        lds_addr,
        (unsigned)(ga & 0xFFFFFFFFu),
        (unsigned)((ga >> 32) & 0x01FFFFFFu)
            | 0x80000000u
    };
    tdm_v8i g1 = {
        (int)((1u << 20) | (3u << 22) | (3u << 25)),
        (int)((K_DIM & 0xFFFFu) << 16),
        (int)((0x8000u) << 16),
        (int)(BK << 16),
        (int)BM,
        (int)K_DIM,
        0,
        0
    };
    tdm_v4i gz4 = {0, 0, 0, 0};
    tdm_v8i gz8 = {0, 0, 0, 0, 0, 0, 0, 0};
    __builtin_amdgcn_tensor_load_to_lds(g0, g1, gz4, gz4, gz8, 0);
}
#endif

__global__ __launch_bounds__(256)
void w8a8_gemm_wmma(const signed char* __restrict__ xp,
                    const signed char* __restrict__ wt,
                    const float* __restrict__ aScale,
                    const float* __restrict__ bScale,
                    float* __restrict__ out) {
    __shared__ __align__(16) signed char As[2][BM * KPAD];
    __shared__ __align__(16) signed char Bs[2][BN * KPAD];

    const int tid    = threadIdx.x;
    const int lane   = tid & 31;
    const int wid    = tid >> 5;
    const int wave_m = wid & 3;
    const int wave_n = wid >> 2;
    const int half   = lane >> 4;
    const int l16    = lane & 15;

    const int m0 = blockIdx.y * BM;
    const int n0 = blockIdx.x * BN;

    const float scale = aScale[0] * bScale[0];

#if USE_TDM
    const signed char* gA0 = xp + (size_t)m0 * K_DIM;
    const signed char* gB0 = wt + (size_t)n0 * K_DIM;
    auto issue = [&](int kt, int buf) {
        if (wid == 0) {
            tdm_issue_tile(gA0 + kt * BK, lds_off(&As[buf][0]));
            tdm_issue_tile(gB0 + kt * BK, lds_off(&Bs[buf][0]));
        }
    };
    auto wait_fill = [&]() {
        if (wid == 0) __builtin_amdgcn_s_wait_tensorcnt(0);
    };
#else
    int row_[2], ch_[2];
#pragma unroll
    for (int j = 0; j < 2; ++j) {
        const int task = j * 256 + tid;
        row_[j] = task >> 2;
        ch_[j]  = (task & 3) * 16;
    }
#if USE_ASYNC_LDS
    auto issue = [&](int kt, int buf) {
#pragma unroll
        for (int j = 0; j < 2; ++j) {
            signed char* ga = const_cast<signed char*>(
                xp + (size_t)(m0 + row_[j]) * K_DIM + kt * BK + ch_[j]);
            signed char* gb = const_cast<signed char*>(
                wt + (size_t)(n0 + row_[j]) * K_DIM + kt * BK + ch_[j]);
            __builtin_amdgcn_global_load_async_to_lds_b128(
                (AS1 v4i_n*)ga,
                (AS3 v4i_n*)&As[buf][row_[j] * KPAD + ch_[j]], 0, 0);
            __builtin_amdgcn_global_load_async_to_lds_b128(
                (AS1 v4i_n*)gb,
                (AS3 v4i_n*)&Bs[buf][row_[j] * KPAD + ch_[j]], 0, 0);
        }
    };
    auto wait_fill = [&]() {
#if __has_builtin(__builtin_amdgcn_s_wait_asynccnt)
        __builtin_amdgcn_s_wait_asynccnt(0);
#else
        asm volatile("s_wait_asynccnt 0x0" ::: "memory");
#endif
    };
#else
    int4 a_st[2], b_st[2];
    auto issue = [&](int kt, int buf) {
#pragma unroll
        for (int j = 0; j < 2; ++j) {
            a_st[j] = *(const int4*)(xp + (size_t)(m0 + row_[j]) * K_DIM
                                        + kt * BK + ch_[j]);
            b_st[j] = *(const int4*)(wt + (size_t)(n0 + row_[j]) * K_DIM
                                        + kt * BK + ch_[j]);
            *(int4*)(&As[buf][row_[j] * KPAD + ch_[j]]) = a_st[j];
            *(int4*)(&Bs[buf][row_[j] * KPAD + ch_[j]]) = b_st[j];
        }
    };
    auto wait_fill = [&]() {};
#endif
#endif

    v8i acc[2][4];
#pragma unroll
    for (int mi = 0; mi < 2; ++mi)
#pragma unroll
        for (int ni = 0; ni < 4; ++ni)
            acc[mi][ni] = (v8i)(0);

    issue(0, 0);
#if !USE_TDM && !USE_ASYNC_LDS
    __syncthreads();
#endif

#pragma unroll 1
    for (int kt = 0; kt < NK; ++kt) {
        const int buf = kt & 1;
        wait_fill();
        __syncthreads();
        if (kt + 1 < NK) issue(kt + 1, buf ^ 1);
#if !USE_TDM && !USE_ASYNC_LDS
        __syncthreads();
#endif

        v8i bf[4];
#pragma unroll
        for (int ni = 0; ni < 4; ++ni) {
            const signed char* bp =
                &Bs[buf][(wave_n * 64 + ni * 16 + l16) * KPAD];
            const int4 lo = *(const int4*)(bp + half * 16);
            const int4 hi = *(const int4*)(bp + 32 + half * 16);
            bf[ni][0] = lo.x; bf[ni][1] = lo.y; bf[ni][2] = lo.z; bf[ni][3] = lo.w;
            bf[ni][4] = hi.x; bf[ni][5] = hi.y; bf[ni][6] = hi.z; bf[ni][7] = hi.w;
        }

#pragma unroll
        for (int mi = 0; mi < 2; ++mi) {
            const signed char* ap =
                &As[buf][(wave_m * 32 + mi * 16 + l16) * KPAD + half * 16];
            v8i af;
#pragma unroll
            for (int c = 0; c < 4; ++c) {
                const int2 v = *(const int2*)(ap + c * 8 + (c >> 1) * 16);
                af[2 * c]     = v.x;
                af[2 * c + 1] = v.y;
            }
#pragma unroll
            for (int ni = 0; ni < 4; ++ni) {
                acc[mi][ni] = __builtin_amdgcn_wmma_i32_16x16x64_iu8(
                    true, af, true, bf[ni], acc[mi][ni], false, false);
            }
        }
    }

#pragma unroll
    for (int mi = 0; mi < 2; ++mi) {
#pragma unroll
        for (int ni = 0; ni < 4; ++ni) {
            const int gn = n0 + wave_n * 64 + ni * 16 + l16;
#pragma unroll
            for (int i = 0; i < 8; ++i) {
                const int gm = m0 + wave_m * 32 + mi * 16 + half * 8 + i;
                float f = (float)acc[mi][ni][i] * scale;
                f = __builtin_rintf(f);
                f = fminf(fmaxf(f, -128.0f), 127.0f);
                out[(size_t)gm * N_DIM + gn] = f;
            }
        }
    }
}

extern "C" void kernel_launch(void* const* d_in, const int* in_sizes, int n_in,
                              void* d_out, int out_size, void* d_ws, size_t ws_size,
                              hipStream_t stream) {
    const int*   x  = (const int*)d_in[0];
    const int*   w  = (const int*)d_in[1];
    const float* a  = (const float*)d_in[2];
    const float* b  = (const float*)d_in[3];
    float*       yo = (float*)d_out;

    signed char* xp = (signed char*)d_ws;
    signed char* wt = xp + XP_BYTES;

    hipLaunchKernelGGL(pack_x_kernel, dim3(8192), dim3(256), 0, stream, x, xp);
    hipLaunchKernelGGL(pack_wt_kernel, dim3(16, 256), dim3(256), 0, stream, w, wt);
    hipLaunchKernelGGL(w8a8_gemm_wmma, dim3(N_DIM / BN, M_TOTAL / BM), dim3(256),
                       0, stream, xp, wt, a, b, yo);
}
